// GNCAE_74474732912750
// MI455X (gfx1250) — matrix operand order repaired and checked by reading, not compiled, not run on hardware
//
#include <hip/hip_runtime.h>
#include <hip/hip_bf16.h>
#include <stdint.h>

// ---------------------------------------------------------------------------
// Types for CDNA5 WMMA (wave32, 16x16x32 bf16 -> f32 accumulate)
// ---------------------------------------------------------------------------
typedef __attribute__((ext_vector_type(16))) __bf16 v16bf;
typedef __attribute__((ext_vector_type(8)))  __bf16 v8bf;
typedef __attribute__((ext_vector_type(4)))  __bf16 v4bf;
typedef __attribute__((ext_vector_type(8)))  float  v8f;
typedef __attribute__((ext_vector_type(4)))  float  v4f;

union AFrag { v16bf v; v8bf h[2]; };

static __device__ __forceinline__ __bf16 f2bf(float f) {
    uint32_t u = __builtin_bit_cast(uint32_t, f);
    uint32_t r = u + 0x7FFFu + ((u >> 16) & 1u);     // round-to-nearest-even
    uint16_t h = (uint16_t)(r >> 16);
    return __builtin_bit_cast(__bf16, h);
}

// Async global->LDS copy of 16 bytes per lane (GLOBAL_LOAD_ASYNC_TO_LDS_B128,
// tracked by ASYNCcnt).  ldsOff is the LDS byte offset (low 32 bits of the
// generic pointer; LDS aperture keeps the offset in addr[31:0]).
static __device__ __forceinline__ void async_copy_b128(uint32_t ldsOff, const void* g) {
    asm volatile("global_load_async_to_lds_b128 %0, %1, off"
                 :: "v"(ldsOff), "v"((uint64_t)(uintptr_t)g) : "memory");
}
static __device__ __forceinline__ uint32_t lds_off(const void* p) {
    return (uint32_t)(uintptr_t)p;
}

// ---------------------------------------------------------------------------
// K1: D[i] = rsqrt( sum_j A[i][j] + 1 )   (+I adds exactly 1 per row)
// ---------------------------------------------------------------------------
__global__ __launch_bounds__(256)
void rowsum_rsqrt_kernel(const float* __restrict__ A, float* __restrict__ D, int N) {
    __shared__ float red[256];
    const int row = blockIdx.x;
    const float* Ar = A + (size_t)row * N;
    float s = 0.f;
    for (int j = threadIdx.x * 4; j < N; j += 256 * 4) {
        const v4f a = __builtin_nontemporal_load((const v4f*)(Ar + j));
        s += a.x + a.y + a.z + a.w;
    }
    red[threadIdx.x] = s;
    __syncthreads();
    for (int off = 128; off > 0; off >>= 1) {
        if ((int)threadIdx.x < off) red[threadIdx.x] += red[threadIdx.x + off];
        __syncthreads();
    }
    if (threadIdx.x == 0) D[row] = rsqrtf(red[0] + 1.0f);
}

// ---------------------------------------------------------------------------
// K2: A_bf[i][j] = bf16( D[i] * (A[i][j] + (i==j)) * D[j] )
// fp32 A is streamed (nontemporal); bf16 A_norm stored RT so the 128MB panel
// can stay resident in the 192MB L2 for the two big GEMMs.
// ---------------------------------------------------------------------------
__global__ __launch_bounds__(256)
void anorm_bf16_kernel(const float* __restrict__ A, const float* __restrict__ D,
                       __bf16* __restrict__ Ab, int N) {
    const int row = blockIdx.y;
    const int j0  = (blockIdx.x * 256 + threadIdx.x) * 4;
    const float di = D[row];
    const v4f a = __builtin_nontemporal_load((const v4f*)(A + (size_t)row * N + j0));
    const float in[4] = {a.x, a.y, a.z, a.w};
    v4bf o;
    #pragma unroll
    for (int t = 0; t < 4; ++t) {
        const int j = j0 + t;
        float x = in[t] + ((j == row) ? 1.0f : 0.0f);
        o[t] = f2bf(di * x * D[j]);
    }
    *(v4bf*)(Ab + (size_t)row * N + j0) = o;
}

// ---------------------------------------------------------------------------
// K3: conversions / zero-fill  (n multiple of 1024)
// ---------------------------------------------------------------------------
__global__ __launch_bounds__(256)
void cvt_bf16_kernel(const float* __restrict__ src, __bf16* __restrict__ dst, int n) {
    const int i = (blockIdx.x * 256 + threadIdx.x) * 4;
    if (i >= n) return;
    const v4f a = *(const v4f*)(src + i);
    v4bf o; o[0] = f2bf(a.x); o[1] = f2bf(a.y); o[2] = f2bf(a.z); o[3] = f2bf(a.w);
    *(v4bf*)(dst + i) = o;
}

__global__ __launch_bounds__(256)
void cvt_bf16_scale_relu_kernel(const float* __restrict__ src, __bf16* __restrict__ dst,
                                int n, float s) {
    const int i = (blockIdx.x * 256 + threadIdx.x) * 4;
    if (i >= n) return;
    const v4f a = *(const v4f*)(src + i);
    v4bf o;
    o[0] = f2bf(fmaxf(s * a.x, 0.f)); o[1] = f2bf(fmaxf(s * a.y, 0.f));
    o[2] = f2bf(fmaxf(s * a.z, 0.f)); o[3] = f2bf(fmaxf(s * a.w, 0.f));
    *(v4bf*)(dst + i) = o;
}

__global__ __launch_bounds__(256)
void zero_f32_kernel(float* __restrict__ p, int n) {
    const int i = (blockIdx.x * 256 + threadIdx.x) * 4;
    if (i >= n) return;
    *(v4f*)(p + i) = v4f{0.f, 0.f, 0.f, 0.f};
}

// ---------------------------------------------------------------------------
// K4: l2-normalize rows of src[R x C], write transposed bf16 dstT[c*R + r]
// ---------------------------------------------------------------------------
__global__ __launch_bounds__(256)
void rownorm_T_kernel(const float* __restrict__ src, __bf16* __restrict__ dstT,
                      int Rows, int Cols) {
    __shared__ float red[256];
    const int row = blockIdx.x;
    const float* xr = src + (size_t)row * Cols;
    float ss = 0.f;
    for (int c = threadIdx.x; c < Cols; c += 256) { const float v = xr[c]; ss += v * v; }
    red[threadIdx.x] = ss;
    __syncthreads();
    for (int off = 128; off > 0; off >>= 1) {
        if ((int)threadIdx.x < off) red[threadIdx.x] += red[threadIdx.x + off];
        __syncthreads();
    }
    const float scale = 1.0f / fmaxf(sqrtf(red[0]), 1e-12f);
    for (int c = threadIdx.x; c < Cols; c += 256)
        dstT[(size_t)c * Rows + row] = f2bf(xr[c] * scale);
}

static __device__ __forceinline__ float epilogue(float x, float alpha, int mode) {
    x *= alpha;
    if (mode == 1)      x = fmaxf(x, 0.0f);
    else if (mode == 2) x = __builtin_amdgcn_rcpf(1.0f + __expf(-x));  // v_rcp_f32
    return x;
}

// ---------------------------------------------------------------------------
// K5a: LDS-staged bf16 NT GEMM (wide-N cases: N multiple of 256).
//   C[M x N] (+)= epilogue( alpha * A[M x K] @ Bt[N x K]^T ), K-chunk = blockIdx.z
//   Block = 256 threads = 8 waves as 4 colgroups x 2 rowgroups.
//   Block tile = 64 rows x 256 cols; wave tile = 32 rows x 64 cols
//   (2 A-frags x 4 B-frags = 8 WMMA per 32-K step).
//   Per 32-K step the block stages A(64x32) and B(256x32) bf16 slices into
//   LDS with GLOBAL_LOAD_ASYNC_TO_LDS_B128 (double buffered, ASYNCcnt +
//   s_wait_asynccnt + barrier), then WMMAs read fragments from LDS in the
//   ISA 7.12.2 per-lane layout:
//     A 16x32 : lane L (m=L&15, hi=L>>4): e[0..7]=A[m][k0+hi*8+..],
//               e[8..15]=A[m][k0+16+hi*8+..]
//     B 32x16 : lane L (n=L&15, hi=L>>4): e[0..7]=Bt[n][k0+hi*8+..], e[8..15]=Bt[n][k0+16+hi*8+..]
//     C 16x16 : vgpr v, lane L: C[row0 + v + 8*hi][col + (L&15)]
// ---------------------------------------------------------------------------
#define LSTRIDE 40   // 32 K-elements + 8 pad (80B rows -> spread LDS banks)

template <bool ATOMIC>
__global__ __launch_bounds__(256)
void gemm_bf16_nt_lds_kernel(const __bf16* __restrict__ A, const __bf16* __restrict__ Bt,
                             float* __restrict__ C, int M, int N, int K, int kChunk,
                             float alpha, int mode, int streamOut) {
    __shared__ __bf16 smA[2][64 * LSTRIDE];    //  2 x  5KB
    __shared__ __bf16 smB[2][256 * LSTRIDE];   //  2 x 20KB

    const int tid  = threadIdx.x;
    const int lane = tid & 31;
    const int wave = tid >> 5;
    const int hi   = lane >> 4;
    const int lm   = lane & 15;
    const int cg   = wave & 3;        // colgroup 0..3
    const int rg   = wave >> 2;       // rowgroup 0..1

    const int rowBase = blockIdx.y * 64;
    const int colBase = blockIdx.x * 256;
    const int kOff    = blockIdx.z * kChunk;
    const int nsteps  = kChunk >> 5;

    // staging roles: thread t copies B col t (4 x b128) and A row t>>2 chunk t&3
    const __bf16* gB = Bt + (size_t)(colBase + tid) * K + kOff;
    const __bf16* gA = A + (size_t)(rowBase + (tid >> 2)) * K + kOff + (tid & 3) * 8;
    const uint32_t lB = lds_off(&smB[0][tid * LSTRIDE]);
    const uint32_t lA = lds_off(&smA[0][(tid >> 2) * LSTRIDE + (tid & 3) * 8]);
    const uint32_t bufStrideB = (uint32_t)(256 * LSTRIDE * sizeof(__bf16));
    const uint32_t bufStrideA = (uint32_t)(64 * LSTRIDE * sizeof(__bf16));

    auto stage = [&](int buf, int s) {
        const __bf16* b = gB + s * 32;
        const uint32_t lb = lB + buf * bufStrideB;
        async_copy_b128(lb +  0, b +  0);
        async_copy_b128(lb + 16, b +  8);
        async_copy_b128(lb + 32, b + 16);
        async_copy_b128(lb + 48, b + 24);
        async_copy_b128(lA + buf * bufStrideA, gA + s * 32);
    };

    v8f acc[2][4] = {{v8f{}, v8f{}, v8f{}, v8f{}}, {v8f{}, v8f{}, v8f{}, v8f{}}};

    stage(0, 0);
    for (int s = 0; s < nsteps; ++s) {
        const int cur = s & 1;
        if (s + 1 < nsteps) {
            stage(cur ^ 1, s + 1);
            asm volatile("s_wait_asynccnt 0x5" ::: "memory");  // oldest 5 (step s) done
        } else {
            asm volatile("s_wait_asynccnt 0x0" ::: "memory");
        }
        __syncthreads();

        AFrag a0, a1;
        {
            const __bf16* pa0 = &smA[cur][(rg * 32 + lm) * LSTRIDE];
            const __bf16* pa1 = &smA[cur][(rg * 32 + 16 + lm) * LSTRIDE];
            a0.h[0] = *(const v8bf*)(pa0 + hi * 8);
            a0.h[1] = *(const v8bf*)(pa0 + 16 + hi * 8);
            a1.h[0] = *(const v8bf*)(pa1 + hi * 8);
            a1.h[1] = *(const v8bf*)(pa1 + 16 + hi * 8);
        }
        #pragma unroll
        for (int t = 0; t < 4; ++t) {
            AFrag b;
            const __bf16* pb = &smB[cur][(cg * 64 + t * 16 + lm) * LSTRIDE];
            b.h[0] = *(const v8bf*)(pb + hi * 8);
            b.h[1] = *(const v8bf*)(pb + 16 + hi * 8);
            acc[0][t] = __builtin_amdgcn_wmma_f32_16x16x32_bf16(false, a0.v, false, b.v, (short)0, acc[0][t], false, false);
            acc[1][t] = __builtin_amdgcn_wmma_f32_16x16x32_bf16(false, a1.v, false, b.v, (short)0, acc[1][t], false, false);
        }
        __syncthreads();   // all waves done reading buf `cur` before its re-stage
    }

    const int row0 = rowBase + rg * 32;
    const int col0 = colBase + cg * 64;
    #pragma unroll
    for (int f = 0; f < 2; ++f) {
        #pragma unroll
        for (int t = 0; t < 4; ++t) {
            const int n = col0 + t * 16 + lm;
            #pragma unroll
            for (int v = 0; v < 8; ++v) {
                const int m = row0 + f * 16 + v + 8 * hi;
                float* dst = C + (size_t)m * N + n;
                if constexpr (ATOMIC) {
                    atomicAdd(dst, acc[f][t][v]);           // global_atomic_add_f32
                } else {
                    const float x = epilogue(acc[f][t][v], alpha, mode);
                    if (streamOut) __builtin_nontemporal_store(x, dst);
                    else           *dst = x;
                }
            }
        }
    }
}

// ---------------------------------------------------------------------------
// K5b: direct-load bf16 NT GEMM for the narrow (N==64) cases.
//   8 waves = 8 rowgroups; block tile 128 rows x 64 cols; K-split atomics.
// ---------------------------------------------------------------------------
template <bool ATOMIC>
__global__ __launch_bounds__(256)
void gemm_bf16_nt_n64_kernel(const __bf16* __restrict__ A, const __bf16* __restrict__ Bt,
                             float* __restrict__ C, int M, int N, int K, int kChunk,
                             float alpha, int mode) {
    const int lane = threadIdx.x & 31;
    const int wave = threadIdx.x >> 5;
    const int hi   = lane >> 4;
    const int lm   = lane & 15;

    const int row0 = (blockIdx.y * 8 + wave) * 16;
    const int kOff = blockIdx.z * kChunk;
    const int kEnd = kOff + kChunk;

    v8f acc[4] = {v8f{}, v8f{}, v8f{}, v8f{}};

    const __bf16* Arow   = A + (size_t)(row0 + lm) * K;
    const __bf16* Bbase0 = Bt + (size_t)(0 * 16 + lm) * K;
    const __bf16* Bbase1 = Bt + (size_t)(1 * 16 + lm) * K;
    const __bf16* Bbase2 = Bt + (size_t)(2 * 16 + lm) * K;
    const __bf16* Bbase3 = Bt + (size_t)(3 * 16 + lm) * K;

    for (int k0 = kOff; k0 < kEnd; k0 += 32) {
        if (k0 + 32 < kEnd) __builtin_prefetch(Arow + k0 + 32, 0, 3);

        AFrag a;
        a.h[0] = *(const v8bf*)(Arow + k0 + hi * 16);
        a.h[1] = *(const v8bf*)(Arow + k0 + hi * 16 + 8);

        const v16bf b0 = *(const v16bf*)(Bbase0 + k0 + hi * 16);
        const v16bf b1 = *(const v16bf*)(Bbase1 + k0 + hi * 16);
        const v16bf b2 = *(const v16bf*)(Bbase2 + k0 + hi * 16);
        const v16bf b3 = *(const v16bf*)(Bbase3 + k0 + hi * 16);

        acc[0] = __builtin_amdgcn_wmma_f32_16x16x32_bf16(false, a.v, false, b0, (short)0, acc[0], false, false);
        acc[1] = __builtin_amdgcn_wmma_f32_16x16x32_bf16(false, a.v, false, b1, (short)0, acc[1], false, false);
        acc[2] = __builtin_amdgcn_wmma_f32_16x16x32_bf16(false, a.v, false, b2, (short)0, acc[2], false, false);
        acc[3] = __builtin_amdgcn_wmma_f32_16x16x32_bf16(false, a.v, false, b3, (short)0, acc[3], false, false);
    }

    #pragma unroll
    for (int t = 0; t < 4; ++t) {
        const int n = t * 16 + lm;
        #pragma unroll
        for (int v = 0; v < 8; ++v) {
            const int m = row0 + v + 8 * hi;
            float* dst = C + (size_t)m * N + n;
            if constexpr (ATOMIC) atomicAdd(dst, acc[t][v]);
            else                  *dst = epilogue(acc[t][v], alpha, mode);
        }
    }
}

// ---------------------------------------------------------------------------
// Host orchestration
// ---------------------------------------------------------------------------
extern "C" void kernel_launch(void* const* d_in, const int* in_sizes, int n_in,
                              void* d_out, int out_size, void* d_ws, size_t ws_size,
                              hipStream_t stream) {
    (void)in_sizes; (void)n_in; (void)out_size; (void)ws_size;
    const int N = 8192, IN = 512, HID = 256, LAT = 64;
    const float S = 1.8f;

    const float* A  = (const float*)d_in[0];
    const float* X  = (const float*)d_in[1];
    const float* W1 = (const float*)d_in[2];
    const float* W2 = (const float*)d_in[3];
    float* out = (float*)d_out;

    char* ws = (char*)d_ws;
    float*  Dv  = (float*)(ws + 0);                        //  32 KB
    __bf16* Ab  = (__bf16*)(ws + 65536);                   // 128 MB  (N*N bf16)
    __bf16* P0  = (__bf16*)(ws + 134283264);               //   8 MB  Xbf -> later Hbf
    float*  P1  = (float*)(ws + 142671872);                //   8 MB  T1  -> later Hraw
    __bf16* P2  = (__bf16*)(ws + 151060480);               //   4 MB  XnT -> later enc_bf
    __bf16* W1b = (__bf16*)(ws + 155254784);               // 256 KB
    __bf16* W2b = (__bf16*)(ws + 155516928);               //  32 KB
    float*  P4  = (float*)(ws + 155549696);                //   2 MB  T2  -> later Mraw
    __bf16* HnT = (__bf16*)(ws + 157646848);               //   1 MB

    // 1) degree vector
    rowsum_rsqrt_kernel<<<N, 256, 0, stream>>>(A, Dv, N);

    // 2) normalized adjacency in bf16
    anorm_bf16_kernel<<<dim3(N / 1024, N), 256, 0, stream>>>(A, Dv, Ab, N);

    // 3) convert inputs to bf16
    cvt_bf16_kernel<<<(N * IN) / 1024, 256, 0, stream>>>(X, P0, N * IN);
    cvt_bf16_kernel<<<(HID * IN) / 1024, 256, 0, stream>>>(W1, W1b, HID * IN);
    cvt_bf16_kernel<<<(LAT * HID) / 1024, 256, 0, stream>>>(W2, W2b, LAT * HID);

    // 4) T1 = X @ W1^T  [N x HID]   (LDS-staged, block 64x256)
    gemm_bf16_nt_lds_kernel<false><<<dim3(1, N / 64, 1), 256, 0, stream>>>(
        P0, W1b, P1, N, HID, IN, IN, 1.0f, 0, 0);

    // 5) XnT = l2norm_rows(T1)^T  [HID x N] bf16
    rownorm_T_kernel<<<N, 256, 0, stream>>>(P1, P2, N, HID);

    // 6) Hraw = A_norm @ Xn  [N x HID]  (K-split x2, atomic accumulate;
    //    relu(S*x) folded into the bf16 conversion below)
    zero_f32_kernel<<<(N * HID) / 1024, 256, 0, stream>>>(P1, N * HID);
    gemm_bf16_nt_lds_kernel<true><<<dim3(1, N / 64, 2), 256, 0, stream>>>(
        Ab, P2, P1, N, HID, N, N / 2, 1.0f, 0, 0);

    // 7) Hbf = bf16(relu(S * Hraw))
    cvt_bf16_scale_relu_kernel<<<(N * HID) / 1024, 256, 0, stream>>>(P1, P0, N * HID, S);

    // 8) T2 = H @ W2^T  [N x LAT]
    gemm_bf16_nt_n64_kernel<false><<<dim3(1, N / 128, 1), 256, 0, stream>>>(
        P0, W2b, P4, N, LAT, HID, HID, 1.0f, 0);

    // 9) HnT = l2norm_rows(T2)^T  [LAT x N] bf16
    rownorm_T_kernel<<<N, 256, 0, stream>>>(P4, HnT, N, LAT);

    // 10) Mraw = A_norm @ Hn  [N x LAT]  (K-split x4; S folds into final alpha)
    zero_f32_kernel<<<(N * LAT) / 1024, 256, 0, stream>>>(P4, N * LAT);
    gemm_bf16_nt_n64_kernel<true><<<dim3(1, N / 128, 4), 256, 0, stream>>>(
        Ab, HnT, P4, N, LAT, N, N / 4, 1.0f, 0);

    // 11) enc_bf = bf16(Mraw)
    cvt_bf16_kernel<<<(N * LAT) / 1024, 256, 0, stream>>>(P4, P2, N * LAT);

    // 12) out = sigmoid(S^2 * (M @ M^T))  [N x N], streaming stores
    gemm_bf16_nt_lds_kernel<false><<<dim3(N / 256, N / 64, 1), 256, 0, stream>>>(
        P2, P2, out, N, N, LAT, LAT, S * S, 2, 1);
}
